// SelfAttention_62088047231060
// MI455X (gfx1250) — matrix operand order repaired and checked by reading, not compiled, not run on hardware
//
#include <hip/hip_runtime.h>
#include <cstdint>
#include <cstddef>

// Problem constants (match reference): B=2,T=16,C=64,A=64,D=32,H=32,W=32
// x: [B,T,C,D,H,W] fp32.  out: same shape fp32.
//
// Pipeline:
//  0) prep_weights: Wk|Wq|Wv fp32 -> contiguous bf16 blob (24KB) in workspace
//  1) xpose_in    : x[B,T,C,D,H,W] -> xt[B,H,W,D,T*C] (bf16), LDS-tiled
//  2) attn        : per (b,h,w) block; weights via TDM tensor_load_to_lds,
//                   X tiles via global_load_async_to_lds_b128 (double-buffered),
//                   WMMA bf16 GEMMs + cross-d softmax
//  3) xpose_out   : featT[B,H,W,D,T*C] (f32) -> out[B,T,C,D,H,W], LDS-tiled

typedef __attribute__((ext_vector_type(16))) __bf16 v16bf;
typedef __attribute__((ext_vector_type(8)))  float  v8f;
typedef __attribute__((ext_vector_type(4)))  unsigned int v4u;
typedef __attribute__((ext_vector_type(8)))  unsigned int v8u;

union Frag16 {
  v16bf v;
  uint4 q[2];
  unsigned short h[16];
};

__device__ __forceinline__ unsigned short f2bf(float f) {
  unsigned u = __float_as_uint(f);
  return (unsigned short)((u + 0x7FFFu + ((u >> 16) & 1u)) >> 16);  // RNE
}
__device__ __forceinline__ unsigned pk2(float a, float b) {
  return (unsigned)f2bf(a) | ((unsigned)f2bf(b) << 16);
}

// A-fragment (16x32 bf16) from a row-major [..][64] LDS tile; row = &M[lane%16][0].
// ISA A layout: lanes 0-15: el0-7 -> K=c0+0..7, el8-15 -> K=c0+16..23; lanes 16-31: +8.
__device__ __forceinline__ v16bf frag_a_row(const unsigned short* row, int c0, int half) {
  Frag16 u;
  u.q[0] = *(const uint4*)(row + c0 + 8 * half);
  u.q[1] = *(const uint4*)(row + c0 + 16 + 8 * half);
  return u.v;
}

// B-fragment (32x16 bf16): column n = lane%16; same K layout as A:
// el0-7 -> K=k0+8*half+0..7, el8-15 -> K=k0+16+8*half+0..7.
__device__ __forceinline__ v16bf frag_b_row(const unsigned short* row, int k0, int half) {
  Frag16 u;
  u.q[0] = *(const uint4*)(row + k0 + 8 * half);
  u.q[1] = *(const uint4*)(row + k0 + 16 + 8 * half);
  return u.v;
}

__device__ __forceinline__ v8f wmma_bf16(v16bf a, v16bf b, v8f c) {
  return __builtin_amdgcn_wmma_f32_16x16x32_bf16(false, a, false, b, (short)0, c,
                                                 false, false);
}

// ---------------------------------------------------------------------------
// Kernel 0: one-shot weight conversion fp32 -> bf16, contiguous [Wk|Wq|Wv].
__global__ void prep_weights_kernel(const float* __restrict__ Wk,
                                    const float* __restrict__ Wq,
                                    const float* __restrict__ Wv,
                                    unsigned short* __restrict__ wbf) {
  int i = blockIdx.x * 256 + threadIdx.x;  // 12288 total
  if (i < 4096)       wbf[i] = f2bf(Wk[i]);
  else if (i < 8192)  wbf[i] = f2bf(Wq[i - 4096]);
  else if (i < 12288) wbf[i] = f2bf(Wv[i - 8192]);
}

// ---------------------------------------------------------------------------
// Kernel 1: x[B,T,C,D,H,W] fp32 -> xt[((b*32+h)*32+w)*32+d][tc] bf16
// Grid: 65536 blocks of (32,8); each block = one 32(tc)x32(w) tile.
__global__ void xpose_in_kernel(const float* __restrict__ x,
                                unsigned short* __restrict__ xt) {
  __shared__ float tile[32][33];
  const int blk = blockIdx.x;
  const int tcb = blk & 31;
  const int h   = (blk >> 5) & 31;
  const int d   = (blk >> 10) & 31;
  const int b   = blk >> 15;
  const int tx = threadIdx.x;  // 0..31
  const int ty = threadIdx.y;  // 0..7
#pragma unroll
  for (int i = 0; i < 4; ++i) {
    int tc = tcb * 32 + ty + i * 8;
    tile[ty + i * 8][tx] =
        x[(size_t)(b * 1024 + tc) * 32768 + d * 1024 + h * 32 + tx];
  }
  __syncthreads();
#pragma unroll
  for (int i = 0; i < 4; ++i) {
    int w = ty + i * 8;
    xt[(size_t)(((b * 32 + h) * 32 + w) * 32 + d) * 1024 + tcb * 32 + tx] =
        f2bf(tile[tx][w]);
  }
}

// ---------------------------------------------------------------------------
// Kernel 3: featT[loc][s*64+c] f32 -> out[B,T,C,D,H,W] f32 (mirror of kernel 1)
__global__ void xpose_out_kernel(const float* __restrict__ featT,
                                 float* __restrict__ out) {
  __shared__ float tile[32][33];
  const int blk = blockIdx.x;
  const int tcb = blk & 31;
  const int h   = (blk >> 5) & 31;
  const int d   = (blk >> 10) & 31;
  const int b   = blk >> 15;
  const int tx = threadIdx.x;
  const int ty = threadIdx.y;
#pragma unroll
  for (int i = 0; i < 4; ++i) {
    int w = ty + i * 8;
    tile[w][tx] =
        featT[(size_t)(((b * 32 + h) * 32 + w) * 32 + d) * 1024 + tcb * 32 + tx];
  }
  __syncthreads();
#pragma unroll
  for (int i = 0; i < 4; ++i) {
    int tc = tcb * 32 + ty + i * 8;
    out[(size_t)(b * 1024 + tc) * 32768 + d * 1024 + h * 32 + tx] =
        tile[tx][ty + i * 8];
  }
}

// ---------------------------------------------------------------------------
// Kernel 2: attention. One block per (b,h,w): 512 threads = 16 waves, each wave
// owns d = wv*2 and wv*2+1. Softmax denominator reduced across all 32 d via LDS.
//
// Dynamic LDS layout (bytes):
//   [0)      W bf16 blob [Wk|Wq|Wv][64][64]          24576  (filled by TDM)
//   [24576)  den f32 [16][16]                        1024
//   [25600)  X  bf16 per-(wave,it) [16][2][16][64]   65536  (async double-buffer)
//   [91136)  K  bf16 per-wave [16][16][64]           32768
//   [123904) Q  bf16 per-wave [16][16][64]           32768
//   [156672) Vt bf16 per-d    [32][64][16]           65536  (Vt[c][t]: t contiguous)
//   total 222208
#define ATTN_SMEM_BYTES 222208

__global__ void __launch_bounds__(512) attn_kernel(
    const unsigned short* __restrict__ xt,
    const unsigned short* __restrict__ wbf,
    const float* __restrict__ bk, const float* __restrict__ bq,
    const float* __restrict__ bv,
    float* __restrict__ featT) {
  extern __shared__ char smem[];
  unsigned short* Wkb  = (unsigned short*)smem;
  unsigned short* Wqb  = Wkb + 4096;
  unsigned short* Wvb  = Wqb + 4096;
  float*          den  = (float*)(smem + 24576);
  unsigned short* Xall = (unsigned short*)(smem + 25600);
  unsigned short* Kall = (unsigned short*)(smem + 91136);
  unsigned short* Qall = (unsigned short*)(smem + 123904);
  unsigned short* Vall = (unsigned short*)(smem + 156672);

  const int tid  = threadIdx.x;
  const int lane = tid & 31;
  const int wv   = tid >> 5;        // 0..15
  const int ln   = lane & 15;
  const int half = lane >> 4;

  const int blk = blockIdx.x;       // 0..2047
  const int w = blk & 31;
  const int h = (blk >> 5) & 31;
  const int b = blk >> 10;

  // --- wave 0: DMA the 24KB bf16 weight blob into LDS via the Tensor Data
  // Mover (2D descriptor, data_size=2B, tile 12288x1), then drain TENSORcnt
  // before joining the barrier so all waves see the data. ---
  if (wv == 0) {
    const unsigned long long ga = (unsigned long long)(size_t)wbf;
    const unsigned ldsw = (unsigned)(size_t)(void*)smem;
    v4u g0;
    g0.x = 1u;                                   // count=1, user descriptor
    g0.y = ldsw;                                 // lds_addr (bytes)
    g0.z = (unsigned)ga;                         // global_addr[31:0]
    g0.w = ((unsigned)(ga >> 32) & 0x01FFFFFFu)  // global_addr[56:32]
           | (2u << 30);                         // type = 2 ("image")
    v8u g1;
    g1.s0 = 1u << 16;       // workgroup_mask=0, data_size=1 (2 bytes)
    g1.s1 = 12288u << 16;   // tensor_dim0[15:0]
    g1.s2 = 1u << 16;       // tensor_dim0[31:16]=0, tensor_dim1=1
    g1.s3 = 12288u << 16;   // tile_dim0 = 12288
    g1.s4 = 1u;             // tile_dim1 = 1, tile_dim2 = 0
    g1.s5 = 12288u;         // tensor_dim0_stride
    g1.s6 = 0u;
    g1.s7 = 0u;
    asm volatile("tensor_load_to_lds %0, %1" ::"s"(g0), "s"(g1) : "memory");
    __builtin_amdgcn_s_wait_tensorcnt(0);
  }
  if (tid < 256) den[tid] = 0.0f;

  // --- every wave: kick off async global->LDS copies of BOTH owned X tiles
  // (2KB each) into the double buffer; consumed behind s_wait_asynccnt. ---
  {
    const unsigned ldsx = (unsigned)(size_t)(void*)Xall;
#pragma unroll
    for (int it = 0; it < 2; ++it) {
      const int d = wv * 2 + it;
      const size_t loc = (size_t)(((b * 32 + h) * 32 + w) * 32 + d);
      const unsigned long long ga =
          (unsigned long long)(size_t)(xt + loc * 1024 + lane * 32);
      const unsigned lds = ldsx + (unsigned)((wv * 2 + it) * 2048 + lane * 64);
#pragma unroll
      for (int i = 0; i < 4; ++i) {
        asm volatile("global_load_async_to_lds_b128 %0, %1, off offset:%2"
                     ::"v"(lds), "v"(ga), "i"(i * 16)
                     : "memory");
      }
    }
  }

  __syncthreads();  // weights in LDS, den cleared

  unsigned short* Kw = Kall + wv * 1024;  // [16][64]
  unsigned short* Qw = Qall + wv * 1024;  // [16][64]

  float e[2][8];  // exp(logits) for this lane's att D-fragment, per owned d

  // ---------------- phase 1: projections + logits, per owned d ----------------
#pragma unroll
  for (int it = 0; it < 2; ++it) {
    const int d = wv * 2 + it;
    unsigned short* Xw = Xall + (wv * 2 + it) * 1024;

    // Async loads complete in order: after 1st tile's 4 -> cnt<=4; all -> 0.
    if (it == 0) asm volatile("s_wait_asynccnt 4" ::: "memory");
    else         asm volatile("s_wait_asynccnt 0" ::: "memory");

    // K and Q projections: D[t][a] = sum_c X[t][c] * W[a][c] + b[a]
#pragma unroll
    for (int n = 0; n < 4; ++n) {
      v8f ck = {};
      v8f cq = {};
      const unsigned short* wkRow = Wkb + (n * 16 + ln) * 64;
      const unsigned short* wqRow = Wqb + (n * 16 + ln) * 64;
#pragma unroll
      for (int kk = 0; kk < 2; ++kk) {
        v16bf a = frag_a_row(Xw + ln * 64, 32 * kk, half);
        ck = wmma_bf16(a, frag_b_row(wkRow, 32 * kk, half), ck);
        cq = wmma_bf16(a, frag_b_row(wqRow, 32 * kk, half), cq);
      }
      const float bbk = bk[n * 16 + ln];
      const float bbq = bq[n * 16 + ln];
#pragma unroll
      for (int r = 0; r < 8; ++r) {
        Kw[(r + 8 * half) * 64 + n * 16 + ln] = f2bf(ck[r] + bbk);
        Qw[(r + 8 * half) * 64 + n * 16 + ln] = f2bf(cq[r] + bbq);
      }
    }

    // V projection, stored transposed Vt[c][t] (t contiguous) for feat B-frags.
    unsigned short* Vd = Vall + d * 1024;
#pragma unroll
    for (int n = 0; n < 4; ++n) {
      v8f cv = {};
      const unsigned short* wvRow = Wvb + (n * 16 + ln) * 64;
#pragma unroll
      for (int kk = 0; kk < 2; ++kk) {
        cv = wmma_bf16(frag_a_row(Xw + ln * 64, 32 * kk, half),
                       frag_b_row(wvRow, 32 * kk, half), cv);
      }
      const float bb = bv[n * 16 + ln];
      uint4 q;
      q.x = pk2(cv[0] + bb, cv[1] + bb);
      q.y = pk2(cv[2] + bb, cv[3] + bb);
      q.z = pk2(cv[4] + bb, cv[5] + bb);
      q.w = pk2(cv[6] + bb, cv[7] + bb);
      // lane holds column c = n*16+ln, rows t = r+8*half -> one b128 store
      *(uint4*)(Vd + (n * 16 + ln) * 16 + 8 * half) = q;
    }

    // att[t][s] = sum_a K[t][a] * Q[s][a]; lane holds s=ln, t = r+8*half.
    v8f ce = {};
#pragma unroll
    for (int kk = 0; kk < 2; ++kk) {
      ce = wmma_bf16(frag_a_row(Kw + ln * 64, 32 * kk, half),
                     frag_b_row(Qw + ln * 64, 32 * kk, half), ce);
    }
#pragma unroll
    for (int r = 0; r < 8; ++r) {
      float ev = __expf(ce[r] * 0.125f);  // scale = 1/sqrt(A=64)
      e[it][r] = ev;
      atomicAdd(&den[(r + 8 * half) * 16 + ln], ev);  // softmax over d axis
    }
  }

  __syncthreads();  // all 32 d-slices accumulated into den

  // ---------------- phase 2: feat = P @ V, per owned d ----------------
#pragma unroll
  for (int it = 0; it < 2; ++it) {
    const int d = wv * 2 + it;
    const size_t loc = (size_t)(((b * 32 + h) * 32 + w) * 32 + d);
    unsigned short* Vd = Vall + d * 1024;

    // P[s][t] = e[t][s]/den[t][s]; lane holds s=ln, t=r+8*half -> exactly the
    // A-fragment positions for the K=16 (zero-padded to 32) feat GEMM.
    float p[8];
#pragma unroll
    for (int r = 0; r < 8; ++r)
      p[r] = e[it][r] / den[(r + 8 * half) * 16 + ln];
    Frag16 pf;
    pf.q[0].x = pk2(p[0], p[1]);
    pf.q[0].y = pk2(p[2], p[3]);
    pf.q[0].z = pk2(p[4], p[5]);
    pf.q[0].w = pk2(p[6], p[7]);
    pf.q[1] = make_uint4(0u, 0u, 0u, 0u);  // K=16..31 padding

    float* outLoc = featT + loc * 1024;
#pragma unroll
    for (int n = 0; n < 4; ++n) {
      Frag16 vf;
      {  // el0-7 carry K = t = 8*half+0..7; el8-15 are padding
        vf.q[0] = *(const uint4*)(Vd + (n * 16 + ln) * 16 + 8 * half);
        vf.q[1] = make_uint4(0u, 0u, 0u, 0u);
      }
      v8f cf = {};
      cf = wmma_bf16(pf.v, vf.v, cf);
#pragma unroll
      for (int r = 0; r < 8; ++r)
        outLoc[(r + 8 * half) * 64 + n * 16 + ln] = cf[r];
    }
  }
}

// ---------------------------------------------------------------------------
extern "C" void kernel_launch(void* const* d_in, const int* in_sizes, int n_in,
                              void* d_out, int out_size, void* d_ws, size_t ws_size,
                              hipStream_t stream) {
  (void)in_sizes; (void)n_in; (void)out_size; (void)ws_size;
  const float* x  = (const float*)d_in[0];
  const float* Wk = (const float*)d_in[1];
  const float* bk = (const float*)d_in[2];
  const float* Wq = (const float*)d_in[3];
  const float* bq = (const float*)d_in[4];
  const float* Wv = (const float*)d_in[5];
  const float* bv = (const float*)d_in[6];
  float* out = (float*)d_out;

  // Workspace: xt bf16 (128MB) + featT f32 (256MB) + weight blob bf16 (24KB)
  unsigned short* xt = (unsigned short*)d_ws;
  float* featT = (float*)((char*)d_ws + (size_t)134217728);
  unsigned short* wbf = (unsigned short*)((char*)d_ws + (size_t)402653184);

  dim3 tb(32, 8);
  prep_weights_kernel<<<48, 256, 0, stream>>>(Wk, Wq, Wv, wbf);
  xpose_in_kernel<<<65536, tb, 0, stream>>>(x, xt);
  attn_kernel<<<2048, 512, ATTN_SMEM_BYTES, stream>>>(xt, wbf, bk, bq, bv, featT);
  xpose_out_kernel<<<65536, tb, 0, stream>>>(featT, out);
}
